// NeuralCDE_13537736917740
// MI455X (gfx1250) — matrix operand order repaired and checked by reading, not compiled, not run on hardware
//
#include <hip/hip_runtime.h>

typedef __attribute__((ext_vector_type(16))) _Float16     v16h;
typedef __attribute__((ext_vector_type(8)))  float        v8f;
typedef __attribute__((ext_vector_type(4)))  unsigned int v4u;
typedef __attribute__((ext_vector_type(8)))  int          v8i_;
typedef __attribute__((ext_vector_type(4)))  int          v4i_;

#define B_      64
#define L_      512
#define C_      8
#define H_      64
#define WID_    128
#define OUTD    (C_*H_)        // 512
#define TSTEPS  (L_-1)         // 511
#define MB      16             // batch rows per block
#define PH_     (TSTEPS * B_ * C_)   // phase stride in g[] (elements)

__device__ __forceinline__ float fast_tanh(float x) {
  x = fminf(15.f, fmaxf(-15.f, x));
  float e = __expf(2.f * x);        // v_exp_f32
  return (e - 1.f) / (e + 1.f);
}

// Hermite spline derivatives -> g[phase][t][b][c], phase in {0:tau=0, 1:tau=.5, 2:tau=1}
__global__ void spline_kernel(const float* __restrict__ X, float* __restrict__ g) {
  int idx = blockIdx.x * blockDim.x + threadIdx.x;
  const int total = B_ * TSTEPS * C_;
  if (idx >= total) return;
  int c = idx & (C_ - 1);
  int t = (idx >> 3) % TSTEPS;
  int b = idx / (TSTEPS * C_);
  float xt  = X[(b * L_ + t) * C_ + c];
  float xt1 = X[(b * L_ + t + 1) * C_ + c];
  float diff = xt1 - xt;
  float dl = (t == 0) ? diff : (xt - X[(b * L_ + t - 1) * C_ + c]);
  float dr = diff;
  float c3 = 3.f * diff - 2.f * dl - dr;
  float e  = -2.f * diff + dl + dr;
  int o = t * (B_ * C_) + b * C_ + c;
  g[o]           = dl;                          // dX0
  g[PH_ + o]     = dl + c3 + 0.75f * e;         // dXh
  g[2 * PH_ + o] = dl + 2.f * c3 + 3.f * e;     // dX1
}

// Issue one TDM 2D tile load: 3 rows (phases, stride PH_) x 128 contiguous floats
// from g[·][t][bstart*C_..] into LDS at lds_off. Tracked by TENSORcnt.
__device__ __forceinline__ void tdm_prefetch_g(const float* __restrict__ gbase,
                                               int t, int bstart, unsigned lds_off) {
  unsigned long long ga =
      (unsigned long long)(const void*)(gbase + (size_t)t * (B_ * C_) + bstart * C_);
  v4u g0;
  g0[0] = 1u;                                          // count=1 (valid descriptor)
  g0[1] = lds_off;                                     // lds_addr (bytes)
  g0[2] = (unsigned)(ga & 0xFFFFFFFFull);              // global_addr[31:0]
  g0[3] = (unsigned)((ga >> 32) & 0x01FFFFFFull)       // global_addr[56:32]
        | (2u << 30);                                  // type=2 ("image")
  v8i_ g1;
  g1[0] = (int)(2u << 16);      // workgroup_mask=0, data_size=2 (4 bytes)
  g1[1] = (int)(128u << 16);    // tensor_dim0[15:0]=128 (bits 63:48)
  g1[2] = (int)(3u << 16);      // tensor_dim0[31:16]=0, tensor_dim1[15:0]=3
  g1[3] = (int)(128u << 16);    // tensor_dim1[31:16]=0, tile_dim0=128
  g1[4] = 3;                    // tile_dim1=3, tile_dim2=0
  g1[5] = (int)PH_;             // tensor_dim0_stride[31:0] = PH_ (elements)
  g1[6] = 0;                    // stride[47:32]=0, tensor_dim1_stride lo=0
  g1[7] = 0;
  v4i_ zz  = {0, 0, 0, 0};      // groups 2/3 unused (2D tensor)
  v8i_ zz8 = {0, 0, 0, 0, 0, 0, 0, 0};
  __builtin_amdgcn_tensor_load_to_lds(g0, g1, zz, zz, zz8, 0);
}

__launch_bounds__(256, 1)
__global__ void cde_kernel(const float* __restrict__ X,
                           const float* __restrict__ Winit, const float* __restrict__ binit,
                           const float* __restrict__ W1, const float* __restrict__ b1,
                           const float* __restrict__ W2, const float* __restrict__ b2,
                           const float* __restrict__ g, float* __restrict__ out) {
  __shared__ float    sz [MB * H_];        // RK4 base state (f32, 4 KB)
  __shared__ _Float16 szh[MB * H_];        // f16 shadow of current stage input (2 KB)
  __shared__ float    sks[MB * H_];        // k-accumulator (4 KB)
  __shared__ _Float16 sh [MB * WID_];      // hidden activations, A of GEMM2 (4 KB)
  __shared__ float    su [MB * OUTD];      // tanh output (32 KB)
  __shared__ float    sg [2][3 * MB * C_]; // ping-pong spline derivs (2 x 1.5 KB)

  const int tid  = threadIdx.x;
  const int wave = tid >> 5;
  const int lane = tid & 31;
  const int n16  = lane & 15;   // tile column (B/C/D) / tile row (A)
  const int hi   = lane >> 4;   // lane-half selector
  const int bstart = blockIdx.x * MB;

  // ---- hoist loop-invariant B fragments + biases into VGPRs (read global once) ----
  v16h bW1[2];
#pragma unroll
  for (int kc = 0; kc < 2; ++kc) {
    const float* src = &W1[(wave * 16 + n16) * H_ + kc * 32 + hi * 16];
#pragma unroll
    for (int r = 0; r < 16; ++r) bW1[kc][r] = (_Float16)src[r];
  }
  v16h bW2[4][4];
#pragma unroll
  for (int jt = 0; jt < 4; ++jt)
#pragma unroll
    for (int kc = 0; kc < 4; ++kc) {
      const float* src = &W2[(wave * 64 + jt * 16 + n16) * WID_ + kc * 32 + hi * 16];
#pragma unroll
      for (int r = 0; r < 16; ++r) bW2[jt][kc][r] = (_Float16)src[r];
    }
  const float bias1 = b1[wave * 16 + n16];
  float bias2[4];
#pragma unroll
  for (int jt = 0; jt < 4; ++jt) bias2[jt] = b2[wave * 64 + jt * 16 + n16];

  // ---- z0 = X[:,0] @ Winit^T + binit (16x64 slice) ----
  for (int idx = tid; idx < MB * H_; idx += 256) {
    int m = idx >> 6, h = idx & 63;
    float acc = binit[h];
#pragma unroll
    for (int c = 0; c < C_; ++c)
      acc += X[(bstart + m) * (L_ * C_) + c] * Winit[h * C_ + c];
    sz[idx]  = acc;
    szh[idx] = (_Float16)acc;
  }

  // ---- TDM prefetch of step-0 spline derivatives ----
  if (wave == 0)
    tdm_prefetch_g(g, 0, bstart, (unsigned)(unsigned long long)(uintptr_t)&sg[0][0]);
  __syncthreads();

  for (int t = 0; t < TSTEPS; ++t) {
    const int cur = t & 1, nxt = cur ^ 1;
    if (wave == 0) {
      __builtin_amdgcn_s_wait_tensorcnt(0);   // step-t tile landed in sg[cur]
    }
    __syncthreads();                          // publish sg[cur] to all waves
    if (wave == 0 && (t + 1) < TSTEPS)        // async prefetch of step t+1
      tdm_prefetch_g(g, t + 1, bstart, (unsigned)(unsigned long long)(uintptr_t)&sg[nxt][0]);

#pragma unroll
    for (int s = 0; s < 4; ++s) {
      // ---- GEMM1: h = relu(zin @ W1^T + b1); 1 N-tile per wave, K=64 ----
      {
        v8f acc;
#pragma unroll
        for (int r = 0; r < 8; ++r) acc[r] = bias1;   // bias folded into C
#pragma unroll
        for (int kc = 0; kc < 2; ++kc) {
          const int k0 = kc * 32;
          v16h a;
#pragma unroll
          for (int j = 0; j < 8; ++j) {               // two 16B runs -> ds_load_b128
            a[j]     = szh[n16 * H_ + k0 + hi * 16 + j];
            a[j + 8] = szh[n16 * H_ + k0 + hi * 16 + 8 + j];
          }
          acc = __builtin_amdgcn_wmma_f32_16x16x32_f16(false, a, false, bW1[kc],
                                                       (short)0, acc, false, false);
        }
#pragma unroll
        for (int r = 0; r < 8; ++r) {                 // D: VGPR r -> row r + 8*hi
          int mm = r + 8 * hi;
          float v = acc[r];
          v = v > 0.f ? v : 0.f;
          sh[mm * WID_ + wave * 16 + n16] = (_Float16)v;
        }
      }
      __syncthreads();

      // ---- GEMM2: u = tanh(h @ W2^T + b2); 4 N-tiles per wave, K=128 ----
      {
        v8f acc[4];
#pragma unroll
        for (int jt = 0; jt < 4; ++jt)
#pragma unroll
          for (int r = 0; r < 8; ++r) acc[jt][r] = bias2[jt];
#pragma unroll
        for (int kc = 0; kc < 4; ++kc) {
          const int k0 = kc * 32;
          v16h a;
#pragma unroll
          for (int j = 0; j < 8; ++j) {               // A fragment reused across N-tiles
            a[j]     = sh[n16 * WID_ + k0 + hi * 16 + j];
            a[j + 8] = sh[n16 * WID_ + k0 + hi * 16 + 8 + j];
          }
#pragma unroll
          for (int jt = 0; jt < 4; ++jt)
            acc[jt] = __builtin_amdgcn_wmma_f32_16x16x32_f16(false, a, false, bW2[jt][kc],
                                                             (short)0, acc[jt], false, false);
        }
#pragma unroll
        for (int jt = 0; jt < 4; ++jt) {
          const int col = wave * 64 + jt * 16 + n16;
#pragma unroll
          for (int r = 0; r < 8; ++r) {
            int mm = r + 8 * hi;
            su[mm * OUTD + col] = fast_tanh(acc[jt][r]);
          }
        }
      }
      __syncthreads();

      // ---- contraction over C=8 + RK4 state update ----
      const int gsel = (s == 0) ? 0 : (s == 3 ? 2 : 1);
      for (int idx = tid; idx < MB * H_; idx += 256) {
        int m = idx >> 6, h = idx & 63;
        const float* gp = &sg[cur][gsel * (MB * C_) + m * C_];
        float kv = 0.f;
#pragma unroll
        for (int i = 0; i < C_; ++i) kv += su[m * OUTD + h * C_ + i] * gp[i];
        if (s == 0) {
          sks[idx] = kv;
          float zv = sz[idx] + 0.5f * kv; szh[idx] = (_Float16)zv;
        } else if (s == 1) {
          sks[idx] += 2.f * kv;
          float zv = sz[idx] + 0.5f * kv; szh[idx] = (_Float16)zv;
        } else if (s == 2) {
          sks[idx] += 2.f * kv;
          float zv = sz[idx] + kv;        szh[idx] = (_Float16)zv;
        } else {
          float zv = sz[idx] + (sks[idx] + kv) * (1.f / 6.f);
          sz[idx] = zv;                   szh[idx] = (_Float16)zv;
        }
      }
      __syncthreads();
    }
  }

  // ---- write zT (B, H) ----
  for (int idx = tid; idx < MB * H_; idx += 256) {
    int m = idx >> 6, h = idx & 63;
    out[(bstart + m) * H_ + h] = sz[idx];
  }
}

extern "C" void kernel_launch(void* const* d_in, const int* in_sizes, int n_in,
                              void* d_out, int out_size, void* d_ws, size_t ws_size,
                              hipStream_t stream) {
  const float* X     = (const float*)d_in[0];
  const float* Winit = (const float*)d_in[1];
  const float* binit = (const float*)d_in[2];
  const float* W1    = (const float*)d_in[3];
  const float* b1    = (const float*)d_in[4];
  const float* W2    = (const float*)d_in[5];
  const float* b2    = (const float*)d_in[6];
  float* out = (float*)d_out;
  float* g   = (float*)d_ws;   // 3 * 511 * 64 * 8 floats ≈ 3 MB

  (void)in_sizes; (void)n_in; (void)out_size; (void)ws_size;

  const int total = B_ * TSTEPS * C_;
  spline_kernel<<<(total + 255) / 256, 256, 0, stream>>>(X, g);
  cde_kernel<<<4, 256, 0, stream>>>(X, Winit, binit, W1, b1, W2, b2, g, out);
}
